// VectorQuantizer_54348516164143
// MI455X (gfx1250) — matrix operand order repaired and checked by reading, not compiled, not run on hardware
//
#include <hip/hip_runtime.h>
#include <hip/hip_bf16.h>

// ---------------------------------------------------------------------------
// Types for CDNA5 WMMA
// ---------------------------------------------------------------------------
typedef __attribute__((ext_vector_type(16))) __bf16 v16bf;
typedef __attribute__((ext_vector_type(8)))  float  v8f;

struct B32x { uint4 a, b; };  // 32 bytes

__device__ __forceinline__ unsigned short f2bf(float f) {
  unsigned u = __builtin_bit_cast(unsigned, f);
  unsigned r = u + 0x7FFFu + ((u >> 16) & 1u);   // round-to-nearest-even
  return (unsigned short)(r >> 16);
}

__device__ __forceinline__ v16bf ld_frag16(const unsigned short* p0,
                                           const unsigned short* p1) {
  B32x t;
  t.a = *(const uint4*)p0;   // elements 0..7
  t.b = *(const uint4*)p1;   // elements 8..15
  return __builtin_bit_cast(v16bf, t);
}

// Sorted-descending top-10 push, fully unrolled (stays in registers).
__device__ __forceinline__ void top10_push(float (&v)[10], int (&ix)[10],
                                           float x, int id) {
  if (x > v[9]) {
    v[9] = x; ix[9] = id;
#pragma unroll
    for (int p = 9; p > 0; --p) {
      if (v[p] > v[p - 1]) {
        float tv = v[p]; v[p] = v[p - 1]; v[p - 1] = tv;
        int   ti = ix[p]; ix[p] = ix[p - 1]; ix[p - 1] = ti;
      }
    }
  }
}

// ---------------------------------------------------------------------------
// Constants
// ---------------------------------------------------------------------------
#define EDIM   768
#define NE     100000
#define NQ     2048
#define CHUNK  400
#define NCH    250     /* 100000 / 400 ; half boundary at chunk 125 */
#define WSZ    589824  /* 768*768 */

// output offsets (floats)
#define O1  0L
#define O2  393216L
#define O3  786432L
#define O4  1179648L
#define O5  1572864L
#define O6  1572865L
#define O7  1572866L
#define O8  1966082L
#define O9  1966083L
#define O10 1966084L
#define O11 2359300L
#define O12 2359301L
#define O13 2359302L
#define O14 2359303L
#define O15 2359304L

// ---------------------------------------------------------------------------
// init: zero bitmap + counter + loss accumulators (contiguous ints)
// ---------------------------------------------------------------------------
__global__ __launch_bounds__(256) void k_init(int* bm) {
  int i = blockIdx.x * 256 + threadIdx.x;
  if (i < NE + 5) bm[i] = 0;   // bitmap[100000] + cnt + accs[4]
}

// ---------------------------------------------------------------------------
// fp32 -> bf16 bulk convert (for weight matrices)
// ---------------------------------------------------------------------------
__global__ __launch_bounds__(256) void k_cvt(const float* __restrict__ src,
                                             unsigned short* __restrict__ dst,
                                             int n) {
  int i = blockIdx.x * 256 + threadIdx.x;
  if (i < n) dst[i] = f2bf(src[i]);
}

// ---------------------------------------------------------------------------
// prep: split z into out1/out2, build GT = [graph_feats ; text_feats]
// ---------------------------------------------------------------------------
__global__ __launch_bounds__(256) void k_prep(const float* __restrict__ z,
                                              const float* __restrict__ textf,
                                              const float* __restrict__ graphf,
                                              float* __restrict__ out,
                                              float* __restrict__ GT) {
  long i = (long)blockIdx.x * 256 + threadIdx.x;
  if (i >= 786432) return;
  long q = i / EDIM, d = i % EDIM;
  if (i < 393216) {
    out[O2 + i] = z[q * 1536 + d];          // z_text_embedding
    out[O1 + i] = z[q * 1536 + 768 + d];    // z_graph_embedding
  }
  GT[i] = (i < 393216) ? graphf[i] : textf[i - 393216];
}

// ---------------------------------------------------------------------------
// row L2-normalize -> bf16; optionally record 1/norm
// ---------------------------------------------------------------------------
__global__ __launch_bounds__(256) void k_norm_rows(const float* __restrict__ src,
                                                   unsigned short* __restrict__ dst,
                                                   float* __restrict__ invn) {
  __shared__ float red[256];
  long row = blockIdx.x;
  const float* s = src + row * EDIM;
  float ss = 0.f;
  for (int d = threadIdx.x; d < EDIM; d += 256) { float x = s[d]; ss += x * x; }
  red[threadIdx.x] = ss;
  __syncthreads();
  for (int o = 128; o > 0; o >>= 1) {
    if (threadIdx.x < o) red[threadIdx.x] += red[threadIdx.x + o];
    __syncthreads();
  }
  float inv = 1.0f / fmaxf(sqrtf(red[0]), 1e-12f);
  for (int d = threadIdx.x; d < EDIM; d += 256)
    dst[row * EDIM + d] = f2bf(s[d] * inv);
  if (invn && threadIdx.x == 0) invn[row] = inv;
}

// ---------------------------------------------------------------------------
// generic  C[M x 768] = A[M x 768] @ W[768 x 768]^T + bias  via bf16 WMMA
// W is pre-converted bf16.  grid = (M/16, 6), block = 128 (4 waves, 2 N-tiles)
// ---------------------------------------------------------------------------
__global__ __launch_bounds__(128) void k_gemm(const float* __restrict__ A, long lda,
                                              const unsigned short* __restrict__ W16,
                                              const float* __restrict__ bias,
                                              float* __restrict__ C, long ldc) {
  __shared__ unsigned short As[16 * EDIM];
  const int  tid = threadIdx.x;
  const long m0  = (long)blockIdx.x * 16;
  for (int i = tid; i < 16 * EDIM; i += 128)
    As[i] = f2bf(A[(m0 + i / EDIM) * lda + (i % EDIM)]);
  __syncthreads();

  const int wv = tid >> 5, ln = tid & 31;
  const int mm = ln & 15, hf = ln >> 4;
  const int nb0 = (blockIdx.y * 8 + wv * 2) * 16;

  v8f acc0 = {}; v8f acc1 = {};
  for (int k0 = 0; k0 < EDIM; k0 += 32) {
    const unsigned short* pa = &As[mm * EDIM + k0 + 8 * hf];
    v16bf a = ld_frag16(pa, pa + 16);
    {
      const unsigned short* p = W16 + (long)(nb0 + mm) * EDIM + k0 + 8 * hf;
      v16bf b = ld_frag16(p, p + 16);
      acc0 = __builtin_amdgcn_wmma_f32_16x16x32_bf16(false, a, false, b,
                                                     (short)0, acc0, false, false);
    }
    {
      const unsigned short* p = W16 + (long)(nb0 + 16 + mm) * EDIM + k0 + 8 * hf;
      v16bf b = ld_frag16(p, p + 16);
      acc1 = __builtin_amdgcn_wmma_f32_16x16x32_bf16(false, a, false, b,
                                                     (short)0, acc1, false, false);
    }
  }
#pragma unroll
  for (int t = 0; t < 2; ++t) {
    int n   = nb0 + 16 * t + mm;
    float bb = bias[n];
    v8f acc = t == 0 ? acc0 : acc1;
#pragma unroll
    for (int r = 0; r < 8; ++r)
      C[(m0 + r + 8 * hf) * ldc + n] = acc[r] + bb;
  }
}

// ---------------------------------------------------------------------------
// scores + per-chunk top-10.  grid=(64 qtiles, 250 chunks), block=160 (5 waves)
// Each block: 32 queries x 400 codebook rows.  Each wave: 5 N-tiles x 2 M-tiles
// (B fragment reused by 2 WMMAs -> halves L2 codebook traffic).
// Q tile (48KB) and score tile (51.2KB) overlay the same LDS buffer.
// ---------------------------------------------------------------------------
__global__ __launch_bounds__(160) void k_scores(const unsigned short* __restrict__ Qn,
                                                const unsigned short* __restrict__ cbn,
                                                float* __restrict__ cval,
                                                int* __restrict__ cidx) {
  const int qt = blockIdx.x, ch = blockIdx.y;
  const int qset = qt >> 4;                     // 16 blocks of 32 queries per set
  if (qset == 2 && ch >= 125) return;           // pt sees rows < 50000
  if (qset == 3 && ch <  125) return;           // pg sees rows >= 50000

  __shared__ char smem[32 * CHUNK * 4];         // 51.2 KB, overlaid
  unsigned short* Qs = (unsigned short*)smem;   // 32*768 bf16 = 48 KB
  float*          Ss = (float*)smem;            // 32*400 f32 = 51.2 KB

  const int tid = threadIdx.x;
  const unsigned short* qsrc = Qn + (long)qt * 32 * EDIM;
  for (int i = tid; i < 32 * EDIM; i += 160) Qs[i] = qsrc[i];
  __syncthreads();

  const int wv = tid >> 5, ln = tid & 31;
  const int mm = ln & 15, hf = ln >> 4;
  const int rowbase = ch * CHUNK;

  v8f acc[10] = {{}, {}, {}, {}, {}, {}, {}, {}, {}, {}};
  for (int k0 = 0; k0 < EDIM; k0 += 32) {
    const unsigned short* pa0 = &Qs[mm * EDIM + k0 + 8 * hf];
    v16bf a0 = ld_frag16(pa0, pa0 + 16);
    const unsigned short* pa1 = pa0 + 16 * EDIM;
    v16bf a1 = ld_frag16(pa1, pa1 + 16);
#pragma unroll
    for (int t = 0; t < 5; ++t) {
      int n = rowbase + (wv * 5 + t) * 16 + mm;
      const unsigned short* p = cbn + (long)n * EDIM + k0 + 8 * hf;
      v16bf b = ld_frag16(p, p + 16);
      acc[2 * t]     = __builtin_amdgcn_wmma_f32_16x16x32_bf16(false, a0, false, b,
                                       (short)0, acc[2 * t],     false, false);
      acc[2 * t + 1] = __builtin_amdgcn_wmma_f32_16x16x32_bf16(false, a1, false, b,
                                       (short)0, acc[2 * t + 1], false, false);
    }
  }
  __syncthreads();   // Q tile dead; reuse LDS for scores
#pragma unroll
  for (int t = 0; t < 5; ++t) {
    int ncol = (wv * 5 + t) * 16 + mm;
#pragma unroll
    for (int r = 0; r < 8; ++r) {
      Ss[(r + 8 * hf) * CHUNK + ncol]        = acc[2 * t][r];
      Ss[(16 + r + 8 * hf) * CHUNK + ncol]   = acc[2 * t + 1][r];
    }
  }
  __syncthreads();

  if (tid < 32) {
    float v[10]; int ix[10];
#pragma unroll
    for (int j = 0; j < 10; ++j) { v[j] = -3.0e38f; ix[j] = 0; }
    const float* row = &Ss[tid * CHUNK];
    for (int j = 0; j < CHUNK; ++j) top10_push(v, ix, row[j], rowbase + j);
    long base = ((long)(qt * 32 + tid) * NCH + ch) * 10;
#pragma unroll
    for (int j = 0; j < 10; ++j) { cval[base + j] = v[j]; cidx[base + j] = ix[j]; }
  }
}

// ---------------------------------------------------------------------------
// merge candidates -> global top-10, softmax, gather z_q, loss accum, idx out
// grid = 2048 (one block per query), block = 256
// ---------------------------------------------------------------------------
__global__ __launch_bounds__(256) void k_merge(const float* __restrict__ cval,
                                               const int* __restrict__ cidx,
                                               const float* __restrict__ Zall,
                                               const float* __restrict__ cb,
                                               const float* __restrict__ invn,
                                               float* __restrict__ out,
                                               int* __restrict__ uidx,
                                               float* __restrict__ accs) {
  const int q = blockIdx.x, tid = threadIdx.x;
  const int set = q >> 9;
  const int c0 = (set == 3) ? 125 : 0;
  const int c1 = (set == 2) ? 125 : NCH;
  const int n  = (c1 - c0) * 10;
  const long base = ((long)q * NCH + c0) * 10;

  __shared__ float lv[32 * 10];
  __shared__ int   li[32 * 10];
  __shared__ float fw[10];
  __shared__ int   fi[10];
  __shared__ float red[256];

  if (tid < 32) {
    float v[10]; int ix[10];
#pragma unroll
    for (int j = 0; j < 10; ++j) { v[j] = -3.0e38f; ix[j] = 0; }
    for (int j = tid; j < n; j += 32)
      top10_push(v, ix, cval[base + j], cidx[base + j]);
#pragma unroll
    for (int j = 0; j < 10; ++j) { lv[tid * 10 + j] = v[j]; li[tid * 10 + j] = ix[j]; }
  }
  __syncthreads();
  if (tid == 0) {
    float v[10]; int ix[10];
#pragma unroll
    for (int j = 0; j < 10; ++j) { v[j] = -3.0e38f; ix[j] = 0; }
    for (int j = 0; j < 320; ++j) top10_push(v, ix, lv[j], li[j]);
    // softmax over neg_vals = 2*s - 2  (constant-shift invariant)
    float mx = v[0], wsum = 0.f, w[10];
#pragma unroll
    for (int k = 0; k < 10; ++k) { w[k] = __expf(2.f * (v[k] - mx)); wsum += w[k]; }
#pragma unroll
    for (int k = 0; k < 10; ++k) {
      fi[k] = ix[k];
      fw[k] = (w[k] / wsum) * invn[ix[k]];   // fold 1/||cb row|| into weight
    }
  }
  __syncthreads();

  if (tid < 10) {
    int g = fi[tid];
    uidx[q * 10 + tid] = (set == 3) ? g - 50000 : g;
  }

  float* dst;
  if      (set == 0) dst = out + O3  + (long)q * EDIM;
  else if (set == 1) dst = out + O4  + (long)(q - 512) * EDIM;
  else if (set == 2) dst = out + O7  + (long)(q - 1024) * EDIM;
  else               dst = out + O10 + (long)(q - 1536) * EDIM;

  const float* zf = Zall + (long)q * EDIM;
  float err = 0.f;
  for (int d = tid; d < EDIM; d += 256) {
    float s = 0.f;
#pragma unroll
    for (int k = 0; k < 10; ++k) s += fw[k] * cb[(long)fi[k] * EDIM + d];
    dst[d] = s;
    float e = s - zf[d];
    err += e * e;
  }
  red[tid] = err;
  __syncthreads();
  for (int o = 128; o > 0; o >>= 1) {
    if (tid < o) red[tid] += red[tid + o];
    __syncthreads();
  }
  if (tid == 0) atomicAdd(&accs[set], red[0]);
}

// ---------------------------------------------------------------------------
// scalar losses
// ---------------------------------------------------------------------------
__global__ void k_scalars(const float* __restrict__ accs, float* __restrict__ out) {
  if (threadIdx.x == 0 && blockIdx.x == 0) {
    float inv = 1.0f / 393216.0f;
    float sv = (accs[0] + accs[1]) * inv;
    out[O5]  = sv;          out[O6]  = 0.25f * sv;
    float t  = accs[2] * inv; out[O8]  = t;  out[O9]  = 0.25f * t;
    float g  = accs[3] * inv; out[O11] = g;  out[O12] = 0.25f * g;
  }
}

// ---------------------------------------------------------------------------
// usage: bitmap scatter with distinct counting; buffer starts all-zero
// ---------------------------------------------------------------------------
__global__ __launch_bounds__(256) void k_scatter(const int* __restrict__ uidx,
                                                 int start, int n,
                                                 int* __restrict__ bm,
                                                 int* __restrict__ cnt, int seed) {
  int i = blockIdx.x * 256 + threadIdx.x;
  if (seed && i == 0) {
    if (atomicExch(&bm[0], 1) == 0) atomicAdd(cnt, 1);   // retained zeros in buf
  }
  if (i < n) {
    int g = uidx[start + i];
    if (atomicExch(&bm[g], 1) == 0) atomicAdd(cnt, 1);
  }
}

__global__ void k_wuse(float* __restrict__ dst, const int* __restrict__ cnt) {
  if (threadIdx.x == 0 && blockIdx.x == 0) dst[0] = (float)(*cnt) * 1e-5f;
}

// ---------------------------------------------------------------------------
// launch
// ---------------------------------------------------------------------------
extern "C" void kernel_launch(void* const* d_in, const int* in_sizes, int n_in,
                              void* d_out, int out_size, void* d_ws, size_t ws_size,
                              hipStream_t stream) {
  const float* z      = (const float*)d_in[0];
  const float* textf  = (const float*)d_in[1];
  const float* graphf = (const float*)d_in[2];
  const float* cb     = (const float*)d_in[4];
  const float* Wv     = (const float*)d_in[5];
  const float* bv     = (const float*)d_in[6];
  const float* Wo     = (const float*)d_in[7];
  const float* bo     = (const float*)d_in[8];
  const float* Wpt    = (const float*)d_in[9];
  const float* bpt    = (const float*)d_in[10];
  const float* Wpg    = (const float*)d_in[11];
  const float* bpg    = (const float*)d_in[12];
  float* out = (float*)d_out;

  char* ws = (char*)d_ws;
  size_t off = 0;
  auto alloc = [&](size_t bytes) {
    size_t o = off;
    off = (off + bytes + 255) & ~(size_t)255;
    return o;
  };
  unsigned short* cbn16 = (unsigned short*)(ws + alloc((size_t)NE * EDIM * 2)); // 153.6 MB (L2-resident)
  float*          invn  = (float*)(ws + alloc((size_t)NE * 4));
  float*          H     = (float*)(ws + alloc((size_t)1024 * EDIM * 4));
  float*          Zall  = (float*)(ws + alloc((size_t)NQ * EDIM * 4));
  unsigned short* Qn    = (unsigned short*)(ws + alloc((size_t)NQ * EDIM * 2));
  float*          GT    = (float*)(ws + alloc((size_t)1024 * EDIM * 4));
  unsigned short* Wb    = (unsigned short*)(ws + alloc((size_t)4 * WSZ * 2));
  float*          cval  = (float*)(ws + alloc((size_t)NQ * NCH * 10 * 4));
  int*            cidx  = (int*)(ws + alloc((size_t)NQ * NCH * 10 * 4));
  int*            uidx  = (int*)(ws + alloc((size_t)NQ * 10 * 4));
  int*            bm    = (int*)(ws + alloc((size_t)(NE + 8) * 4));
  int*   cnt  = bm + NE;
  float* accs = (float*)(bm + NE + 1);

  unsigned short* Wv16  = Wb;
  unsigned short* Wo16  = Wb + WSZ;
  unsigned short* Wpt16 = Wb + 2 * (size_t)WSZ;
  unsigned short* Wpg16 = Wb + 3 * (size_t)WSZ;

  k_init<<<(NE + 5 + 255) / 256, 256, 0, stream>>>(bm);
  k_prep<<<3072, 256, 0, stream>>>(z, textf, graphf, out, GT);
  k_norm_rows<<<NE, 256, 0, stream>>>(cb, cbn16, invn);

  // pre-convert weights to bf16 (removes per-k-step cvt VALU from GEMM loop)
  k_cvt<<<(WSZ + 255) / 256, 256, 0, stream>>>(Wv,  Wv16,  WSZ);
  k_cvt<<<(WSZ + 255) / 256, 256, 0, stream>>>(Wo,  Wo16,  WSZ);
  k_cvt<<<(WSZ + 255) / 256, 256, 0, stream>>>(Wpt, Wpt16, WSZ);
  k_cvt<<<(WSZ + 255) / 256, 256, 0, stream>>>(Wpg, Wpg16, WSZ);

  // projections: H = GT @ Wv^T + bv ; Zall[0:1024] = H @ Wo^T + bo
  k_gemm<<<dim3(64, 6), 128, 0, stream>>>(GT, EDIM, Wv16, bv, H, EDIM);
  k_gemm<<<dim3(64, 6), 128, 0, stream>>>(H, EDIM, Wo16, bo, Zall, EDIM);
  // pt / pg from z halves
  k_gemm<<<dim3(32, 6), 128, 0, stream>>>(z,       1536, Wpt16, bpt, Zall + (size_t)1024 * EDIM, EDIM);
  k_gemm<<<dim3(32, 6), 128, 0, stream>>>(z + 768, 1536, Wpg16, bpg, Zall + (size_t)1536 * EDIM, EDIM);

  k_norm_rows<<<NQ, 256, 0, stream>>>(Zall, Qn, nullptr);

  k_scores<<<dim3(64, NCH), 160, 0, stream>>>(Qn, cbn16, cval, cidx);
  k_merge<<<NQ, 256, 0, stream>>>(cval, cidx, Zall, cb, invn, out, uidx, accs);
  k_scalars<<<1, 1, 0, stream>>>(accs, out);

  // usage, sequential: shared (idx_t & idx_g) -> text -> graph
  k_scatter<<<(10240 + 255) / 256, 256, 0, stream>>>(uidx, 0, 10240, bm, cnt, 1);
  k_wuse<<<1, 1, 0, stream>>>(out + O13, cnt);
  k_scatter<<<(5120 + 255) / 256, 256, 0, stream>>>(uidx, 10240, 5120, bm, cnt, 0);
  k_wuse<<<1, 1, 0, stream>>>(out + O14, cnt);
  k_scatter<<<(5120 + 255) / 256, 256, 0, stream>>>(uidx, 15360, 5120, bm, cnt, 0);
  k_wuse<<<1, 1, 0, stream>>>(out + O15, cnt);
}
